// MultiHeadSelfAttention_39797166965449
// MI455X (gfx1250) — matrix operand order repaired and checked by reading, not compiled, not run on hardware
//
#include <hip/hip_runtime.h>

// ---------------------------------------------------------------------------
// MultiHeadSelfAttention for MI455X (gfx1250, wave32, WMMA)
//   B=4, S=2048, D=1024, H=16, Dh=64. ~137 GFLOP -> compute bound -> bf16 WMMA
//   (v_wmma_f32_16x16x32_bf16, f32 accumulate).
// ---------------------------------------------------------------------------

#define D_MODEL 1024
#define SEQ     2048
#define NHEAD   16
#define HDIM    64
#define NROWS   8192              // B*S
#define LDSP    40                // ushort row stride: 80B = 5*16B (b128-aligned)

typedef __attribute__((ext_vector_type(16))) __bf16 v16bf;
typedef __attribute__((ext_vector_type(8)))  float  v8f;

union FragBF { v16bf v; uint4 q4[2]; unsigned u[8]; };

// f32 -> bf16 (round-half-up: differs from RNE only on exact ties)
__device__ __forceinline__ unsigned short f2bf(float f) {
  return (unsigned short)((__float_as_uint(f) + 0x8000u) >> 16);
}
// two f32 -> packed bf16x2 via one v_perm_b32 (bytes {hi.b3,hi.b2,lo.b3,lo.b2})
__device__ __forceinline__ unsigned pack2bf(float lo, float hi) {
  return __builtin_amdgcn_perm(__float_as_uint(hi) + 0x8000u,
                               __float_as_uint(lo) + 0x8000u, 0x07060302u);
}

// 16-bit A-fragment (ISA 7.12.2): lane holds row m = lane%16;
// VGPR0..3 <-> K pairs {0..7}+8*half, VGPR4..7 <-> {16..23}+8*half.
// In a row-major K-contiguous row these are two contiguous 16B chunks.
__device__ __forceinline__ void load_afrag(FragBF& a, const unsigned short* row, int half) {
  a.q4[0] = *(const uint4*)&row[8 * half];
  a.q4[1] = *(const uint4*)&row[16 + 8 * half];
}

// 16-bit B-fragment: lane holds column n = lane%16; same K order as the
// A-fragment: elements 0..7 <-> k = 8*half + e, elements 8..15 <-> 16 + 8*half + (e-8).
// From a row-major B^T row these are two contiguous 16B chunks.
__device__ __forceinline__ void load_bfrag(FragBF& b, const unsigned short* row, int half) {
  b.q4[0] = *(const uint4*)&row[8 * half];
  b.q4[1] = *(const uint4*)&row[16 + 8 * half];
}

// stage one float4 as 4 bf16 into an LDS row (8B store)
__device__ __forceinline__ void stage4(unsigned short* dst, const float4 f) {
  *(uint2*)dst = make_uint2(pack2bf(f.x, f.y), pack2bf(f.z, f.w));
}

// ---------------------------------------------------------------------------
// Kernel 1: fused QKV projection.  C[8192,1024] = X * W^T, cvt to bf16.
//   Block: 256 thr / 8 waves -> 64x128 C tile; wave (mg,ng) = 32x32 subtile,
//   4 WMMA per K-step (2 A-frags x 2 B-frags).  Global->register double
//   buffering overlaps next tile's loads with the WMMA phase.
//   blockIdx.z selects W; Q/K stored [B,H,S,Dh], V transposed [B,H,Dh,S].
// ---------------------------------------------------------------------------
__global__ __launch_bounds__(256) void qkv_proj_kernel(
    const float* __restrict__ x,
    const float* __restrict__ Wq, const float* __restrict__ Wk,
    const float* __restrict__ Wv,
    unsigned short* __restrict__ qw, unsigned short* __restrict__ kw,
    unsigned short* __restrict__ vw)
{
  __shared__ unsigned short a_lds[64 * LDSP];
  __shared__ unsigned short b_lds[128 * LDSP];

  const int tid  = threadIdx.x;
  const int wave = tid >> 5;
  const int lane = tid & 31;
  const int half = lane >> 4;
  const int m    = lane & 15;
  const int mg   = wave >> 2;           // 0..1 : 32-row group
  const int ng   = wave & 3;            // 0..3 : 32-col group
  const int m0   = blockIdx.x * 64;     // 128 row tiles of 64
  const int nblk = blockIdx.y * 128;    // 8 col blocks of 128
  const int z    = blockIdx.z;          // 0=Q 1=K 2=V
  const float* W = (z == 0) ? Wq : (z == 1) ? Wk : Wv;

  // loop-invariant staging coordinates: thread -> (row, 4-col chunk)
  const int sr = tid >> 3;              // 0..31
  const int sc = (tid & 7) * 4;         // 0,4,..,28
  const float* xp0 = &x[(size_t)(m0 + sr) * D_MODEL + sc];
  const float* wp0 = &W[(size_t)(nblk + sr) * D_MODEL + sc];

  // preload K-step 0 into registers
  float4 fa[2], fb[4];
  #pragma unroll
  for (int i = 0; i < 2; ++i) fa[i] = *(const float4*)(xp0 + (size_t)i * 32 * D_MODEL);
  #pragma unroll
  for (int i = 0; i < 4; ++i) fb[i] = *(const float4*)(wp0 + (size_t)i * 32 * D_MODEL);

  v8f accv[2][2] = {};
  for (int kk = 0; kk < D_MODEL; kk += 32) {
    __syncthreads();
    #pragma unroll
    for (int i = 0; i < 2; ++i) stage4(&a_lds[(sr + i * 32) * LDSP + sc], fa[i]);
    #pragma unroll
    for (int i = 0; i < 4; ++i) stage4(&b_lds[(sr + i * 32) * LDSP + sc], fb[i]);
    __syncthreads();

    if (kk + 32 < D_MODEL) {            // issue next tile's loads (in flight
      #pragma unroll                    // across the whole WMMA phase)
      for (int i = 0; i < 2; ++i)
        fa[i] = *(const float4*)(xp0 + (size_t)i * 32 * D_MODEL + kk + 32);
      #pragma unroll
      for (int i = 0; i < 4; ++i)
        fb[i] = *(const float4*)(wp0 + (size_t)i * 32 * D_MODEL + kk + 32);
    }

    FragBF a0, a1, b0, b1;
    load_afrag(a0, (const unsigned short*)&a_lds[(mg * 32 + m) * LDSP], half);
    load_afrag(a1, (const unsigned short*)&a_lds[(mg * 32 + 16 + m) * LDSP], half);
    load_bfrag(b0, (const unsigned short*)&b_lds[(ng * 32 + m) * LDSP], half);
    load_bfrag(b1, (const unsigned short*)&b_lds[(ng * 32 + 16 + m) * LDSP], half);
    accv[0][0] = __builtin_amdgcn_wmma_f32_16x16x32_bf16(false, a0.v, false, b0.v,
                                                         (short)0, accv[0][0], false, false);
    accv[0][1] = __builtin_amdgcn_wmma_f32_16x16x32_bf16(false, a0.v, false, b1.v,
                                                         (short)0, accv[0][1], false, false);
    accv[1][0] = __builtin_amdgcn_wmma_f32_16x16x32_bf16(false, a1.v, false, b0.v,
                                                         (short)0, accv[1][0], false, false);
    accv[1][1] = __builtin_amdgcn_wmma_f32_16x16x32_bf16(false, a1.v, false, b1.v,
                                                         (short)0, accv[1][1], false, false);
  }

  if (z == 2) {                         // V^T [bh][dh][s]
    #pragma unroll
    for (int j = 0; j < 2; ++j) {
      int col = nblk + ng * 32 + j * 16 + m;   // N = lane % 16
      int h = col >> 6, dh = col & 63;
      #pragma unroll
      for (int i = 0; i < 2; ++i) {
        #pragma unroll
        for (int r = 0; r < 8; ++r) {
          int row = m0 + mg * 32 + i * 16 + r + 8 * half;  // M = r + 8*half
          int b_ = row >> 11, s = row & 2047;
          vw[((size_t)(b_ * NHEAD + h) * HDIM + dh) * SEQ + s] = f2bf(accv[i][j][r]);
        }
      }
    }
  } else {                              // Q/K [bh][s][dh]
    unsigned short* __restrict__ dst = (z == 0) ? qw : kw;
    #pragma unroll
    for (int j = 0; j < 2; ++j) {
      int col = nblk + ng * 32 + j * 16 + m;
      int h = col >> 6, dh = col & 63;
      #pragma unroll
      for (int i = 0; i < 2; ++i) {
        #pragma unroll
        for (int r = 0; r < 8; ++r) {
          int row = m0 + mg * 32 + i * 16 + r + 8 * half;
          int b_ = row >> 11, s = row & 2047;
          dst[((size_t)(b_ * NHEAD + h) * SEQ + s) * HDIM + dh] = f2bf(accv[i][j][r]);
        }
      }
    }
  }
}

// ---------------------------------------------------------------------------
// Kernel 2: causal flash attention. One wave per (bh, 16-query tile).
//   Key blocks of 32; scores via 4 WMMA, PV via 4 WMMA; online softmax with
//   16-lane shuffle reductions; P round-trips through per-wave LDS tile.
// ---------------------------------------------------------------------------
__global__ __launch_bounds__(128) void attn_kernel(
    const unsigned short* __restrict__ qw,
    const unsigned short* __restrict__ kw,
    const unsigned short* __restrict__ vw,
    unsigned short* __restrict__ ow)
{
  __shared__ unsigned short p_lds[4 * 16 * LDSP];   // per-wave P tiles
  const int tid  = threadIdx.x;
  const int wave = tid >> 5;
  const int lane = tid & 31;
  const int half = lane >> 4;
  const int m    = lane & 15;
  const int qt   = blockIdx.x;          // 128 query tiles
  const int q0   = qt * 16;
  const int bh   = blockIdx.y * 4 + wave;   // 64 (b,h) pairs

  const unsigned short* qp = qw + (size_t)bh * SEQ * HDIM;
  const unsigned short* kp = kw + (size_t)bh * SEQ * HDIM;
  const unsigned short* vp = vw + (size_t)bh * HDIM * SEQ;   // [dh][s]
  unsigned short* plw = p_lds + wave * 16 * LDSP;

  // Q A-fragments (held in registers for the whole key loop)
  FragBF aq0, aq1;
  {
    const unsigned short* qrow = qp + (size_t)(q0 + m) * HDIM;
    load_afrag(aq0, qrow, half);
    load_afrag(aq1, qrow + 32, half);
  }

  float mrun[8], lrun[8];
  #pragma unroll
  for (int r = 0; r < 8; ++r) { mrun[r] = -1e30f; lrun[r] = 0.f; }
  v8f accv[4] = {};
  const float scale = 0.125f;           // 1/sqrt(64)
  const int nkb = (qt >> 1) + 1;        // causal key-block count

  for (int kb = 0; kb < nkb; ++kb) {
    const int kbase = kb * 32;

    // prefetch next key block's K rows and V^T rows
    if (kb + 1 < nkb) {
      __builtin_prefetch(kp + (size_t)(kbase + 32 + m) * HDIM, 0, 1);
      __builtin_prefetch(vp + (size_t)(lane) * SEQ + kbase + 32, 0, 1);
    }

    // ---- scores: two 16x16 tiles over keys [kbase,kbase+16), [+16,+32)
    v8f sc0 = {}, sc1 = {};
    {
      FragBF bk0, bk1;
      const unsigned short* krow = kp + (size_t)(kbase + m) * HDIM;
      load_bfrag(bk0, krow, half);          // d-range [0,32)
      load_bfrag(bk1, krow + 32, half);     // d-range [32,64)
      sc0 = __builtin_amdgcn_wmma_f32_16x16x32_bf16(false, aq0.v, false, bk0.v,
                                                    (short)0, sc0, false, false);
      sc0 = __builtin_amdgcn_wmma_f32_16x16x32_bf16(false, aq1.v, false, bk1.v,
                                                    (short)0, sc0, false, false);
    }
    {
      FragBF bk0, bk1;
      const unsigned short* krow = kp + (size_t)(kbase + 16 + m) * HDIM;
      load_bfrag(bk0, krow, half);
      load_bfrag(bk1, krow + 32, half);
      sc1 = __builtin_amdgcn_wmma_f32_16x16x32_bf16(false, aq0.v, false, bk0.v,
                                                    (short)0, sc1, false, false);
      sc1 = __builtin_amdgcn_wmma_f32_16x16x32_bf16(false, aq1.v, false, bk1.v,
                                                    (short)0, sc1, false, false);
    }

    // ---- causal mask + online softmax (row M = r + 8*half spans one 16-lane half)
    #pragma unroll
    for (int r = 0; r < 8; ++r) {
      int qi = q0 + r + 8 * half;
      float s0 = ((kbase + m)      <= qi) ? sc0[r] * scale : -1e30f;
      float s1 = ((kbase + 16 + m) <= qi) ? sc1[r] * scale : -1e30f;
      float mx = fmaxf(s0, s1);
      #pragma unroll
      for (int o = 8; o >= 1; o >>= 1) mx = fmaxf(mx, __shfl_xor(mx, o, 32));
      float mn   = fmaxf(mrun[r], mx);
      float corr = __expf(mrun[r] - mn);
      float e0   = __expf(s0 - mn);
      float e1   = __expf(s1 - mn);
      float rs   = e0 + e1;
      #pragma unroll
      for (int o = 8; o >= 1; o >>= 1) rs += __shfl_xor(rs, o, 32);
      lrun[r] = lrun[r] * corr + rs;
      mrun[r] = mn;
      accv[0][r] *= corr; accv[1][r] *= corr;
      accv[2][r] *= corr; accv[3][r] *= corr;
      int prow = r + 8 * half;
      plw[prow * LDSP + m]      = f2bf(e0);
      plw[prow * LDSP + 16 + m] = f2bf(e1);
    }

    // wave-internal DS ordering: stores above -> loads below (same wave only;
    // each wave owns its own p_lds slice, so no block barrier needed)
    asm volatile("s_wait_dscnt 0" ::: "memory");

    // ---- P as A-fragment, then O += P * V  (V^T rows are contiguous in keys)
    FragBF ap;
    load_afrag(ap, (const unsigned short*)&plw[m * LDSP], half);
    #pragma unroll
    for (int t = 0; t < 4; ++t) {
      FragBF bv;
      const unsigned short* vrow = vp + (size_t)(t * 16 + m) * SEQ + kbase;
      load_bfrag(bv, vrow, half);
      accv[t] = __builtin_amdgcn_wmma_f32_16x16x32_bf16(false, ap.v, false, bv.v,
                                                        (short)0, accv[t], false, false);
    }
  }

  // ---- normalize and store O as bf16 in [B,S,D]
  const int b_ = bh >> 4, h = bh & 15;
  #pragma unroll
  for (int r = 0; r < 8; ++r) {
    float inv = 1.0f / lrun[r];
    int q = q0 + r + 8 * half;
    #pragma unroll
    for (int t = 0; t < 4; ++t) {
      int dcol = h * HDIM + t * 16 + m;
      ow[((size_t)b_ * SEQ + q) * D_MODEL + dcol] = f2bf(accv[t][r] * inv);
    }
  }
}

// ---------------------------------------------------------------------------
// Kernel 3: output projection.  out[8192,1024] f32 = O_bf16 * Wo^T.
//   Block: 64x128 tile, 4 WMMA per wave per K-step; A-fragments load directly
//   from bf16 O; Wo staged through LDS with register double buffering.
// ---------------------------------------------------------------------------
__global__ __launch_bounds__(256) void out_proj_kernel(
    const unsigned short* __restrict__ ow,
    const float* __restrict__ Wo,
    float* __restrict__ out)
{
  __shared__ unsigned short b_lds[128 * LDSP];
  const int tid  = threadIdx.x;
  const int wave = tid >> 5;
  const int lane = tid & 31;
  const int half = lane >> 4;
  const int m    = lane & 15;
  const int mg   = wave >> 2;
  const int ng   = wave & 3;
  const int m0   = blockIdx.x * 64;
  const int nblk = blockIdx.y * 128;
  const int sr   = tid >> 3;
  const int sc   = (tid & 7) * 4;
  const float* wp0 = &Wo[(size_t)(nblk + sr) * D_MODEL + sc];

  float4 fb[4];
  #pragma unroll
  for (int i = 0; i < 4; ++i) fb[i] = *(const float4*)(wp0 + (size_t)i * 32 * D_MODEL);

  v8f accv[2][2] = {};
  for (int kk = 0; kk < D_MODEL; kk += 32) {
    __syncthreads();
    #pragma unroll
    for (int i = 0; i < 4; ++i) stage4(&b_lds[(sr + i * 32) * LDSP + sc], fb[i]);
    __syncthreads();

    if (kk + 32 < D_MODEL) {
      #pragma unroll
      for (int i = 0; i < 4; ++i)
        fb[i] = *(const float4*)(wp0 + (size_t)i * 32 * D_MODEL + kk + 32);
      __builtin_prefetch(&ow[(size_t)(m0 + (tid >> 2)) * D_MODEL + kk + 32], 0, 1);
    }

    FragBF a0, a1, b0, b1;
    load_afrag(a0, ow + (size_t)(m0 + mg * 32 + m) * D_MODEL + kk, half);
    load_afrag(a1, ow + (size_t)(m0 + mg * 32 + 16 + m) * D_MODEL + kk, half);
    load_bfrag(b0, (const unsigned short*)&b_lds[(ng * 32 + m) * LDSP], half);
    load_bfrag(b1, (const unsigned short*)&b_lds[(ng * 32 + 16 + m) * LDSP], half);
    accv[0][0] = __builtin_amdgcn_wmma_f32_16x16x32_bf16(false, a0.v, false, b0.v,
                                                         (short)0, accv[0][0], false, false);
    accv[0][1] = __builtin_amdgcn_wmma_f32_16x16x32_bf16(false, a0.v, false, b1.v,
                                                         (short)0, accv[0][1], false, false);
    accv[1][0] = __builtin_amdgcn_wmma_f32_16x16x32_bf16(false, a1.v, false, b0.v,
                                                         (short)0, accv[1][0], false, false);
    accv[1][1] = __builtin_amdgcn_wmma_f32_16x16x32_bf16(false, a1.v, false, b1.v,
                                                         (short)0, accv[1][1], false, false);
  }

  #pragma unroll
  for (int j = 0; j < 2; ++j) {
    int col = nblk + ng * 32 + j * 16 + m;
    #pragma unroll
    for (int i = 0; i < 2; ++i) {
      #pragma unroll
      for (int r = 0; r < 8; ++r) {
        int row = m0 + mg * 32 + i * 16 + r + 8 * half;
        out[(size_t)row * D_MODEL + col] = accv[i][j][r];
      }
    }
  }
}

// ---------------------------------------------------------------------------
extern "C" void kernel_launch(void* const* d_in, const int* in_sizes, int n_in,
                              void* d_out, int out_size, void* d_ws, size_t ws_size,
                              hipStream_t stream) {
  (void)in_sizes; (void)n_in; (void)out_size; (void)ws_size;
  const float* x  = (const float*)d_in[0];
  const float* Wq = (const float*)d_in[1];
  const float* Wk = (const float*)d_in[2];
  const float* Wv = (const float*)d_in[3];
  const float* Wo = (const float*)d_in[4];

  unsigned short* ws = (unsigned short*)d_ws;
  const size_t NQKV = (size_t)4 * NHEAD * SEQ * HDIM;   // 8,388,608 bf16 elems each
  unsigned short* qw = ws;
  unsigned short* kw = ws + NQKV;
  unsigned short* vw = ws + 2 * NQKV;
  unsigned short* o_ = ws + 3 * NQKV;

  qkv_proj_kernel<<<dim3(NROWS / 64, D_MODEL / 128, 3), 256, 0, stream>>>(
      x, Wq, Wk, Wv, qw, kw, vw);
  attn_kernel<<<dim3(SEQ / 16, (4 * NHEAD) / 4), 128, 0, stream>>>(
      qw, kw, vw, o_);
  out_proj_kernel<<<dim3(NROWS / 64, D_MODEL / 128), 256, 0, stream>>>(
      o_, Wo, (float*)d_out);
}
